// DecoderBlock_81690277970442
// MI455X (gfx1250) — matrix operand order repaired and checked by reading, not compiled, not run on hardware
//
#include <hip/hip_runtime.h>
#include <hip/hip_bf16.h>

typedef __bf16 bf16_t;
typedef __attribute__((ext_vector_type(16))) __bf16 v16bf;
typedef __attribute__((ext_vector_type(8)))  __bf16 v8bf;
typedef __attribute__((ext_vector_type(8)))  float  v8f;

union V16 { v16bf v; v8bf h[2]; };

__device__ __forceinline__ v8f wmma_bf16(const v16bf& a, const v16bf& b, const v8f& c) {
    return __builtin_amdgcn_wmma_f32_16x16x32_bf16(false, a, false, b, (short)0, c, false, false);
}

// ---------------- f32 -> bf16 convert ----------------
__global__ __launch_bounds__(256) void cvt_k(const float* __restrict__ in,
                                             bf16_t* __restrict__ out, size_t n) {
    size_t i = (size_t)blockIdx.x * 256 + threadIdx.x;
    size_t stride = (size_t)gridDim.x * 256;
    for (; i < n; i += stride) out[i] = (bf16_t)in[i];
}

// ---------------- LayerNorm (row per block) -> bf16 ----------------
__global__ __launch_bounds__(256) void ln_k(const float* __restrict__ X,
                                            const float* __restrict__ g,
                                            const float* __restrict__ be,
                                            bf16_t* __restrict__ Y, int Dm) {
    __shared__ float red0[8], red1[8];
    int row = blockIdx.x;
    const float* x = X + (size_t)row * Dm;
    float s = 0.f, s2 = 0.f;
    for (int i = threadIdx.x; i < Dm; i += 256) { float v = x[i]; s += v; s2 += v * v; }
    for (int o = 16; o; o >>= 1) { s += __shfl_down(s, o, 32); s2 += __shfl_down(s2, o, 32); }
    int wave = threadIdx.x >> 5, lane = threadIdx.x & 31;
    if (lane == 0) { red0[wave] = s; red1[wave] = s2; }
    __syncthreads();
    if (threadIdx.x == 0) {
        float a = 0.f, b = 0.f;
        for (int w = 0; w < 8; ++w) { a += red0[w]; b += red1[w]; }
        red0[0] = a; red1[0] = b;
    }
    __syncthreads();
    float mean = red0[0] / Dm;
    float var  = red1[0] / Dm - mean * mean;
    float rstd = rsqrtf(var + 1e-5f);
    for (int i = threadIdx.x; i < Dm; i += 256)
        Y[(size_t)row * Dm + i] = (bf16_t)((x[i] - mean) * rstd * g[i] + be[i]);
}

// ---------------- Generic bf16 WMMA GEMM (pipelined) ----------------
// C[M,N] = act( A[M,K](bf16) x B[K,N](bf16) + bias[N] ) (+ residual[M,N])
// Expert mode (counts != null): blockIdx.z = expert, per-expert M=counts[e],
// weight at B + e*wstride, bias at bias + e*biasStride, rows offset by offsets[e].
// rowtok != null: A row gather (rowtok[globalRow] -> row of A), hoisted out of k-loop.
#define BM 128
#define BN 128
#define BK 32
#define LP 40   // padded LDS row stride (bf16 elems): 80B, 16B aligned, conflict-free

__global__ __launch_bounds__(256)
void gemm_bf16_k(const bf16_t* __restrict__ A, const bf16_t* __restrict__ Bw,
                 const float* __restrict__ bias, const float* __restrict__ resid,
                 float* __restrict__ outF, bf16_t* __restrict__ outB,
                 int M, int N, int K, int lda, int act,
                 const int* __restrict__ counts, const int* __restrict__ offsets,
                 const int* __restrict__ rowtok,
                 unsigned long long wstride, int biasStride)
{
    __shared__ __align__(16) bf16_t As[BM][LP];
    __shared__ __align__(16) bf16_t Bs[BN][LP];

    int e = blockIdx.z;
    int Me = M, base = 0;
    const bf16_t* Bp = Bw;
    const float* biasp = bias;
    if (counts) {
        Me = counts[e]; base = offsets[e];
        Bp = Bw + (size_t)e * wstride;
        if (bias) biasp = bias + (size_t)e * biasStride;
    }
    int mtile = blockIdx.x * BM;
    if (mtile >= Me) return;
    int ntile = blockIdx.y * BN;

    int tid = threadIdx.x;
    int lane = tid & 31;
    int wave = tid >> 5;
    int wm = wave & 3;     // 4 waves along M (32 rows each)
    int wn = wave >> 2;    // 2 waves along N (64 cols each)
    int hl = lane & 15, hh = lane >> 4;

    // hoisted staging addresses: 2 A chunks + 2 B chunks of 8 bf16 per thread
    const bf16_t* asrc[2];
    bf16_t*       adst[2];
    const bf16_t* bsrc[2];
    bf16_t*       bdst[2];
#pragma unroll
    for (int i = 0; i < 2; ++i) {
        int c = tid + i * 256;
        {
            int r = c >> 2, kc = (c & 3) * 8;
            int gr = mtile + r;
            int ar = (gr < Me) ? (rowtok ? rowtok[base + gr] : base + gr)
                               : (rowtok ? rowtok[base]      : base);
            asrc[i] = A + (size_t)ar * lda + kc;
            adst[i] = &As[r][kc];
        }
        {
            int k = c >> 4, nc = (c & 15) * 8;
            bsrc[i] = Bp + (size_t)k * N + ntile + nc;
            bdst[i] = &Bs[nc][k];
        }
    }

    v8f acc[2][4];
#pragma unroll
    for (int i = 0; i < 2; ++i)
#pragma unroll
        for (int j = 0; j < 4; ++j) acc[i][j] = (v8f){0.f,0.f,0.f,0.f,0.f,0.f,0.f,0.f};

    // prologue: prefetch k-tile 0
    v8bf aReg[2], bReg[2];
#pragma unroll
    for (int i = 0; i < 2; ++i) {
        aReg[i] = *(const v8bf*)(asrc[i]);
        bReg[i] = *(const v8bf*)(bsrc[i]);
    }

    for (int kt = 0; kt < K; kt += BK) {
        __syncthreads();   // previous tile fully consumed
#pragma unroll
        for (int i = 0; i < 2; ++i) {
            *(v8bf*)adst[i] = aReg[i];
#pragma unroll
            for (int j = 0; j < 8; ++j) bdst[i][(size_t)j * LP] = bReg[i][j];
        }
        __syncthreads();   // tile visible
        if (kt + BK < K) { // prefetch next tile while WMMAs run
#pragma unroll
            for (int i = 0; i < 2; ++i) {
                aReg[i] = *(const v8bf*)(asrc[i] + (kt + BK));
                bReg[i] = *(const v8bf*)(bsrc[i] + (size_t)(kt + BK) * N);
            }
        }

        V16 af[2];
#pragma unroll
        for (int i = 0; i < 2; ++i) {
            int row = wm * 32 + i * 16 + hl;
            int kb = hh * 8;
            af[i].h[0] = *(const v8bf*)&As[row][kb];
            af[i].h[1] = *(const v8bf*)&As[row][kb + 16];
        }
        V16 bfr[4];
#pragma unroll
        for (int j = 0; j < 4; ++j) {
            int col = wn * 64 + j * 16 + hl;
            int kb = hh * 8;
            bfr[j].h[0] = *(const v8bf*)&Bs[col][kb];
            bfr[j].h[1] = *(const v8bf*)&Bs[col][kb + 16];
        }
#pragma unroll
        for (int i = 0; i < 2; ++i)
#pragma unroll
            for (int j = 0; j < 4; ++j)
                acc[i][j] = wmma_bf16(af[i].v, bfr[j].v, acc[i][j]);
    }

#pragma unroll
    for (int i = 0; i < 2; ++i) {
#pragma unroll
        for (int j = 0; j < 4; ++j) {
            int col = ntile + wn * 64 + j * 16 + hl;
            float bv = biasp ? biasp[col] : 0.f;
#pragma unroll
            for (int r = 0; r < 8; ++r) {
                int row = mtile + wm * 32 + i * 16 + r + hh * 8;
                if (row < Me) {
                    float v = acc[i][j][r] + bv;
                    if (act == 1) v = fmaxf(v, 0.f);
                    size_t off = (size_t)(base + row) * N + col;
                    if (resid) v += resid[off];
                    if (outF) outF[off] = v;
                    else      outB[off] = (bf16_t)v;
                }
            }
        }
    }
}

// ---------------- Flash attention (WMMA QK^T + online softmax + WMMA PV) ----------------
#define DHD 64
__global__ __launch_bounds__(256)
void attn_k(const bf16_t* __restrict__ Q, const bf16_t* __restrict__ Kg,
            const bf16_t* __restrict__ Vg, bf16_t* __restrict__ O,
            int Tq, int Sk, int Dm, int causal)
{
    __shared__ __align__(16) bf16_t Vst[64][LP];     // transposed V chunk: [n][k]
    __shared__ float Scm[8][16][33];                 // per-wave score tile
    __shared__ __align__(16) bf16_t Pb[8][16][LP];   // per-wave P tile (bf16)
    __shared__ float Fs[8][16];
    __shared__ float Li[8][16];

    int b = blockIdx.z, h = blockIdx.y;
    int t0b = blockIdx.x * 128;
    int tid = threadIdx.x, lane = tid & 31, wave = tid >> 5;
    int hl = lane & 15, hh = lane >> 4;
    int t0w = t0b + wave * 16;

    const size_t qbase = (size_t)(b * Tq) * Dm + h * DHD;
    const size_t kbase = (size_t)(b * Sk) * Dm + h * DHD;

    V16 q0, q1;
    {
        int row = t0w + hl;
        int kb = hh * 8;
        const bf16_t* qp = Q + qbase + (size_t)row * Dm;
        q0.h[0] = *(const v8bf*)&qp[kb];
        q0.h[1] = *(const v8bf*)&qp[kb + 16];
        q1.h[0] = *(const v8bf*)&qp[32 + kb];
        q1.h[1] = *(const v8bf*)&qp[32 + kb + 16];
    }

    v8f oacc[4];
#pragma unroll
    for (int j = 0; j < 4; ++j) oacc[j] = (v8f){0.f,0.f,0.f,0.f,0.f,0.f,0.f,0.f};
    float mrun = -3e38f, lrun = 0.f;   // live in lanes 0..15
    const float scale = 0.125f;        // 1/sqrt(64)

    int smax = causal ? ((t0b + 128 < Sk) ? t0b + 128 : Sk) : Sk;

    // hoisted staging addresses + chunk-0 prefetch
    const int vk = tid >> 3, vnc = (tid & 7) * 8;            // V staging coords
    const bf16_t* vsrc = Vg + kbase + (size_t)vk * Dm + vnc; // + s0*Dm per chunk
    const bf16_t* ksrc = Kg + kbase + (size_t)hl * Dm;       // + (s0 + j*16)*Dm
    const int kkb = hh * 8;

    v8bf vReg = *(const v8bf*)(vsrc);
    V16 kfr[2][2];
#pragma unroll
    for (int j = 0; j < 2; ++j) {
        const bf16_t* kp = ksrc + (size_t)(j * 16) * Dm;
        kfr[j][0].h[0] = *(const v8bf*)&kp[kkb];
        kfr[j][0].h[1] = *(const v8bf*)&kp[kkb + 16];
        kfr[j][1].h[0] = *(const v8bf*)&kp[32 + kkb];
        kfr[j][1].h[1] = *(const v8bf*)&kp[32 + kkb + 16];
    }

    for (int s0 = 0; s0 < smax; s0 += 32) {
        bool hasNext = (s0 + 32) < smax;
        __syncthreads();
        {   // commit prefetched V chunk transposed into LDS
#pragma unroll
            for (int j = 0; j < 8; ++j) Vst[vnc + j][vk] = vReg[j];
        }
        __syncthreads();
        if (hasNext) vReg = *(const v8bf*)(vsrc + (size_t)(s0 + 32) * Dm);

        // scores: two 16x16 subtiles, each = 2 WMMAs over dh (K frags prefetched)
#pragma unroll
        for (int j = 0; j < 2; ++j) {
            v8f cfr = (v8f){0.f,0.f,0.f,0.f,0.f,0.f,0.f,0.f};
            cfr = wmma_bf16(q0.v, kfr[j][0].v, cfr);
            cfr = wmma_bf16(q1.v, kfr[j][1].v, cfr);
#pragma unroll
            for (int r = 0; r < 8; ++r)
                Scm[wave][r + hh * 8][j * 16 + hl] = cfr[r] * scale;
        }
        if (hasNext) {   // prefetch next chunk's K fragments during softmax/PV
#pragma unroll
            for (int j = 0; j < 2; ++j) {
                const bf16_t* kp = ksrc + (size_t)(s0 + 32 + j * 16) * Dm;
                kfr[j][0].h[0] = *(const v8bf*)&kp[kkb];
                kfr[j][0].h[1] = *(const v8bf*)&kp[kkb + 16];
                kfr[j][1].h[0] = *(const v8bf*)&kp[32 + kkb];
                kfr[j][1].h[1] = *(const v8bf*)&kp[32 + kkb + 16];
            }
        }
        __syncthreads();

        // per-row online softmax: lanes 0..15 own one row each
        if (lane < 16) {
            int r = lane;
            int t = t0w + r;
            float* srow = Scm[wave][r];
            float vals[32];
            float cmax = -3e38f;
#pragma unroll
            for (int cc = 0; cc < 32; ++cc) {
                float v = srow[cc];
                if (causal && (s0 + cc > t)) v = -3e38f;
                vals[cc] = v;
                cmax = fmaxf(cmax, v);
            }
            if (cmax > -1e30f) {
                float mnew = fmaxf(mrun, cmax);
                float fsv = __expf(mrun - mnew);
                float sum = 0.f;
#pragma unroll
                for (int cc = 0; cc < 32; ++cc) {
                    float pv = (vals[cc] > -1e30f) ? __expf(vals[cc] - mnew) : 0.f;
                    Pb[wave][r][cc] = (bf16_t)pv;
                    sum += pv;
                }
                lrun = lrun * fsv + sum;
                mrun = mnew;
                Fs[wave][r] = fsv;
            } else {
#pragma unroll
                for (int cc = 0; cc < 32; ++cc) Pb[wave][r][cc] = (bf16_t)0.f;
                Fs[wave][r] = 1.f;
            }
        }
        __syncthreads();

        // rescale accumulators and do P x V
        float fsc[8];
#pragma unroll
        for (int r = 0; r < 8; ++r) fsc[r] = Fs[wave][r + hh * 8];
#pragma unroll
        for (int j = 0; j < 4; ++j)
#pragma unroll
            for (int r = 0; r < 8; ++r) oacc[j][r] *= fsc[r];

        V16 pa;
        {
            int kb = hh * 8;
            pa.h[0] = *(const v8bf*)&Pb[wave][hl][kb];
            pa.h[1] = *(const v8bf*)&Pb[wave][hl][kb + 16];
        }
#pragma unroll
        for (int j = 0; j < 4; ++j) {
            int col = j * 16 + hl;
            int kb = hh * 8;
            V16 vb;
            vb.h[0] = *(const v8bf*)&Vst[col][kb];
            vb.h[1] = *(const v8bf*)&Vst[col][kb + 16];
            oacc[j] = wmma_bf16(pa.v, vb.v, oacc[j]);
        }
    }

    if (lane < 16) Li[wave][lane] = 1.0f / lrun;
    __syncthreads();
    float liv[8];
#pragma unroll
    for (int r = 0; r < 8; ++r) liv[r] = Li[wave][r + hh * 8];
#pragma unroll
    for (int j = 0; j < 4; ++j) {
        int col = h * DHD + j * 16 + hl;
#pragma unroll
        for (int r = 0; r < 8; ++r) {
            int t = t0w + r + hh * 8;
            O[(size_t)(b * Tq + t) * Dm + col] = (bf16_t)(oacc[j][r] * liv[r]);
        }
    }
}

// ---------------- MoE gating: softmax + top-2 per token (one wave/token) ----------------
__global__ __launch_bounds__(256)
void gate_k(const bf16_t* __restrict__ X, const bf16_t* __restrict__ Wg,
            const float* __restrict__ bg, int* counts,
            int* tokexp, int* tokpos, float* tokgate, int Ntok)
{
    int wave = threadIdx.x >> 5, lane = threadIdx.x & 31;
    int tok = blockIdx.x * 8 + wave;
    if (tok >= Ntok) return;
    const bf16_t* x = X + (size_t)tok * 1024;
    float a[8];
#pragma unroll
    for (int e = 0; e < 8; ++e) a[e] = 0.f;
    for (int i = 0; i < 32; ++i) {
        int k = lane * 32 + i;
        float xv = (float)x[k];
#pragma unroll
        for (int e = 0; e < 8; ++e) a[e] += xv * (float)Wg[k * 8 + e];
    }
    for (int o = 16; o; o >>= 1)
#pragma unroll
        for (int e = 0; e < 8; ++e) a[e] += __shfl_down(a[e], o, 32);
    if (lane == 0) {
        float mx = -3e38f;
#pragma unroll
        for (int e = 0; e < 8; ++e) { a[e] += bg[e]; mx = fmaxf(mx, a[e]); }
        float sum = 0.f;
#pragma unroll
        for (int e = 0; e < 8; ++e) { a[e] = __expf(a[e] - mx); sum += a[e]; }
        int e0 = 0;
#pragma unroll
        for (int e = 1; e < 8; ++e) if (a[e] > a[e0]) e0 = e;
        int e1 = (e0 == 0) ? 1 : 0;
#pragma unroll
        for (int e = 0; e < 8; ++e) if (e != e0 && a[e] > a[e1]) e1 = e;
        float p0 = a[e0], p1 = a[e1], tt = p0 + p1;
        int pos0 = atomicAdd(&counts[e0], 1);
        int pos1 = atomicAdd(&counts[e1], 1);
        tokexp[tok * 2] = e0;  tokexp[tok * 2 + 1] = e1;
        tokpos[tok * 2] = pos0; tokpos[tok * 2 + 1] = pos1;
        tokgate[tok * 2] = p0 / tt; tokgate[tok * 2 + 1] = p1 / tt;
    }
}

__global__ void zero_k(int* p, int n) {
    int i = blockIdx.x * 256 + threadIdx.x;
    if (i < n) p[i] = 0;
}

__global__ void prefix_k(const int* counts, int* offsets) {
    if (threadIdx.x == 0 && blockIdx.x == 0) {
        int s = 0;
        for (int e = 0; e < 8; ++e) { offsets[e] = s; s += counts[e]; }
        offsets[8] = s;
    }
}

__global__ void map_k(const int* __restrict__ tokexp, const int* __restrict__ tokpos,
                      const int* __restrict__ offsets, int* __restrict__ rowtok,
                      int* __restrict__ tokrow, int Nent) {
    int i = blockIdx.x * 256 + threadIdx.x;
    if (i >= Nent) return;
    int row = offsets[tokexp[i]] + tokpos[i];
    rowtok[row] = i >> 1;
    tokrow[i] = row;
}

__global__ __launch_bounds__(256)
void combine_k(const float* __restrict__ h2, const float* __restrict__ y,
               const int* __restrict__ tokrow, const float* __restrict__ tokgate,
               float* __restrict__ out, int Dm)
{
    int tok = blockIdx.x;
    int r0 = tokrow[tok * 2], r1 = tokrow[tok * 2 + 1];
    float g0 = tokgate[tok * 2], g1 = tokgate[tok * 2 + 1];
    for (int c = threadIdx.x; c < Dm; c += 256) {
        out[(size_t)tok * Dm + c] = h2[(size_t)tok * Dm + c]
            + g0 * y[(size_t)r0 * Dm + c] + g1 * y[(size_t)r1 * Dm + c];
    }
}

// ---------------- host orchestration ----------------
static inline void cvt(const float* in, bf16_t* out, size_t n, hipStream_t s) {
    int grid = (int)((n + 255) / 256);
    if (grid > 65535) grid = 65535;
    cvt_k<<<grid, 256, 0, s>>>(in, out, n);
}

extern "C" void kernel_launch(void* const* d_in, const int* in_sizes, int n_in,
                              void* d_out, int out_size, void* d_ws, size_t ws_size,
                              hipStream_t stream)
{
    (void)in_sizes; (void)n_in; (void)out_size; (void)ws_size;
    constexpr int Bc = 4, Tc = 1024, Sc = 1024, Dc = 1024, Hc = 16, Ec = 8, FFc = 4096;
    constexpr int Ntok = Bc * Tc;          // 4096 decoder tokens
    constexpr int Nenc = Bc * Sc;          // 4096 encoder tokens
    constexpr int Nent = Ntok * 2;         // token-expert entries

    const float* x    = (const float*)d_in[0];
    const float* enc  = (const float*)d_in[1];
    const float* ln1g = (const float*)d_in[4];  const float* ln1b = (const float*)d_in[5];
    const float* ln2g = (const float*)d_in[6];  const float* ln2b = (const float*)d_in[7];
    const float* ln3g = (const float*)d_in[8];  const float* ln3b = (const float*)d_in[9];
    const float* wq1 = (const float*)d_in[10]; const float* bq1 = (const float*)d_in[11];
    const float* wk1 = (const float*)d_in[12]; const float* bk1 = (const float*)d_in[13];
    const float* wv1 = (const float*)d_in[14]; const float* bv1 = (const float*)d_in[15];
    const float* wo1 = (const float*)d_in[16]; const float* bo1 = (const float*)d_in[17];
    const float* wq2 = (const float*)d_in[18]; const float* bq2 = (const float*)d_in[19];
    const float* wk2 = (const float*)d_in[20]; const float* bk2 = (const float*)d_in[21];
    const float* wv2 = (const float*)d_in[22]; const float* bv2 = (const float*)d_in[23];
    const float* wo2 = (const float*)d_in[24]; const float* bo2 = (const float*)d_in[25];
    const float* wg  = (const float*)d_in[26]; const float* bg  = (const float*)d_in[27];
    const float* w1  = (const float*)d_in[28]; const float* b1  = (const float*)d_in[29];
    const float* w2  = (const float*)d_in[30]; const float* b2  = (const float*)d_in[31];
    float* out = (float*)d_out;

    char* p = (char*)d_ws;
    auto alloc = [&](size_t bytes) -> char* {
        char* r = p; p += (bytes + 255) & ~(size_t)255; return r;
    };
    const size_t DD = (size_t)Dc * Dc;
    bf16_t* wq1b = (bf16_t*)alloc(DD * 2); bf16_t* wk1b = (bf16_t*)alloc(DD * 2);
    bf16_t* wv1b = (bf16_t*)alloc(DD * 2); bf16_t* wo1b = (bf16_t*)alloc(DD * 2);
    bf16_t* wq2b = (bf16_t*)alloc(DD * 2); bf16_t* wk2b = (bf16_t*)alloc(DD * 2);
    bf16_t* wv2b = (bf16_t*)alloc(DD * 2); bf16_t* wo2b = (bf16_t*)alloc(DD * 2);
    bf16_t* wgb  = (bf16_t*)alloc((size_t)Dc * Ec * 2);
    bf16_t* w1b  = (bf16_t*)alloc((size_t)Ec * Dc * FFc * 2);
    bf16_t* w2b  = (bf16_t*)alloc((size_t)Ec * FFc * Dc * 2);
    bf16_t* nb   = (bf16_t*)alloc((size_t)Ntok * Dc * 2);   // LN output (reused per layer)
    bf16_t* encb = (bf16_t*)alloc((size_t)Nenc * Dc * 2);
    bf16_t* qb   = (bf16_t*)alloc((size_t)Ntok * Dc * 2);
    bf16_t* kb   = (bf16_t*)alloc((size_t)Nenc * Dc * 2);
    bf16_t* vb   = (bf16_t*)alloc((size_t)Nenc * Dc * 2);
    bf16_t* attb = (bf16_t*)alloc((size_t)Ntok * Dc * 2);
    float*  h1   = (float*)alloc((size_t)Ntok * Dc * 4);
    float*  h2   = (float*)alloc((size_t)Ntok * Dc * 4);
    bf16_t* hbuf = (bf16_t*)alloc((size_t)Nent * FFc * 2);
    float*  ybuf = (float*)alloc((size_t)Nent * Dc * 4);
    int* counts  = (int*)alloc(16 * 4);
    int* offsets = (int*)alloc(16 * 4);
    int* tokexp  = (int*)alloc((size_t)Nent * 4);
    int* tokpos  = (int*)alloc((size_t)Nent * 4);
    float* tokgate = (float*)alloc((size_t)Nent * 4);
    int* rowtok  = (int*)alloc((size_t)Nent * 4);
    int* tokrow  = (int*)alloc((size_t)Nent * 4);

    // weights -> bf16
    cvt(wq1, wq1b, DD, stream); cvt(wk1, wk1b, DD, stream);
    cvt(wv1, wv1b, DD, stream); cvt(wo1, wo1b, DD, stream);
    cvt(wq2, wq2b, DD, stream); cvt(wk2, wk2b, DD, stream);
    cvt(wv2, wv2b, DD, stream); cvt(wo2, wo2b, DD, stream);
    cvt(wg,  wgb,  (size_t)Dc * Ec, stream);
    cvt(w1,  w1b,  (size_t)Ec * Dc * FFc, stream);
    cvt(w2,  w2b,  (size_t)Ec * FFc * Dc, stream);
    cvt(enc, encb, (size_t)Nenc * Dc, stream);

    dim3 blk(256);
    dim3 gProj(Ntok / BM, Dc / BN, 1);        // 32 x 8

    // ---- sublayer 1: self attention ----
    ln_k<<<Ntok, blk, 0, stream>>>(x, ln1g, ln1b, nb, Dc);
    gemm_bf16_k<<<gProj, blk, 0, stream>>>(nb, wq1b, bq1, nullptr, nullptr, qb,
        Ntok, Dc, Dc, Dc, 0, nullptr, nullptr, nullptr, 0ull, 0);
    gemm_bf16_k<<<gProj, blk, 0, stream>>>(nb, wk1b, bk1, nullptr, nullptr, kb,
        Ntok, Dc, Dc, Dc, 0, nullptr, nullptr, nullptr, 0ull, 0);
    gemm_bf16_k<<<gProj, blk, 0, stream>>>(nb, wv1b, bv1, nullptr, nullptr, vb,
        Ntok, Dc, Dc, Dc, 0, nullptr, nullptr, nullptr, 0ull, 0);
    attn_k<<<dim3(Tc / 128, Hc, Bc), blk, 0, stream>>>(qb, kb, vb, attb, Tc, Tc, Dc, 1);
    gemm_bf16_k<<<gProj, blk, 0, stream>>>(attb, wo1b, bo1, x, h1, nullptr,
        Ntok, Dc, Dc, Dc, 0, nullptr, nullptr, nullptr, 0ull, 0);

    // ---- sublayer 2: cross attention ----
    ln_k<<<Ntok, blk, 0, stream>>>(h1, ln2g, ln2b, nb, Dc);
    gemm_bf16_k<<<gProj, blk, 0, stream>>>(nb, wq2b, bq2, nullptr, nullptr, qb,
        Ntok, Dc, Dc, Dc, 0, nullptr, nullptr, nullptr, 0ull, 0);
    gemm_bf16_k<<<gProj, blk, 0, stream>>>(encb, wk2b, bk2, nullptr, nullptr, kb,
        Nenc, Dc, Dc, Dc, 0, nullptr, nullptr, nullptr, 0ull, 0);
    gemm_bf16_k<<<gProj, blk, 0, stream>>>(encb, wv2b, bv2, nullptr, nullptr, vb,
        Nenc, Dc, Dc, Dc, 0, nullptr, nullptr, nullptr, 0ull, 0);
    attn_k<<<dim3(Tc / 128, Hc, Bc), blk, 0, stream>>>(qb, kb, vb, attb, Tc, Sc, Dc, 0);
    gemm_bf16_k<<<gProj, blk, 0, stream>>>(attb, wo2b, bo2, h1, h2, nullptr,
        Ntok, Dc, Dc, Dc, 0, nullptr, nullptr, nullptr, 0ull, 0);

    // ---- sublayer 3: routed top-2 MoE ----
    ln_k<<<Ntok, blk, 0, stream>>>(h2, ln3g, ln3b, nb, Dc);
    zero_k<<<1, 256, 0, stream>>>(counts, 16);
    gate_k<<<Ntok / 8, blk, 0, stream>>>(nb, wgb, bg, counts, tokexp, tokpos, tokgate, Ntok);
    prefix_k<<<1, 1, 0, stream>>>(counts, offsets);
    map_k<<<(Nent + 255) / 256, blk, 0, stream>>>(tokexp, tokpos, offsets, rowtok, tokrow, Nent);
    // expert up-proj: gathered rows, ReLU, bf16 out  (cap M=Ntok per expert)
    gemm_bf16_k<<<dim3(Ntok / BM, FFc / BN, Ec), blk, 0, stream>>>(
        nb, w1b, b1, nullptr, nullptr, hbuf,
        Ntok, FFc, Dc, Dc, 1, counts, offsets, rowtok,
        (unsigned long long)Dc * FFc, FFc);
    // expert down-proj: segment rows, f32 out
    gemm_bf16_k<<<dim3(Ntok / BM, Dc / BN, Ec), blk, 0, stream>>>(
        hbuf, w2b, b2, nullptr, ybuf, nullptr,
        Ntok, Dc, FFc, FFc, 0, counts, offsets, nullptr,
        (unsigned long long)FFc * Dc, Dc);
    combine_k<<<Ntok, blk, 0, stream>>>(h2, ybuf, tokrow, tokgate, out, Dc);
}
